// NSAP_lp_80410377716358
// MI455X (gfx1250) — matrix operand order repaired and checked by reading, not compiled, not run on hardware
//
#include <hip/hip_runtime.h>

typedef __attribute__((ext_vector_type(16))) __bf16          v16bf;
typedef __attribute__((ext_vector_type(16))) unsigned short  v16us;
typedef __attribute__((ext_vector_type(8)))  float           v8f;

// Split fp32 into bf16 hi (truncation, exact residual) + bf16 lo (truncation).
// Pure bit ops: no dependence on f32->bf16 cvt lowering.
__device__ __forceinline__ void split_bf16(float x, unsigned short& h, unsigned short& l) {
    unsigned u  = __float_as_uint(x);
    float    hf = __uint_as_float(u & 0xFFFF0000u);
    h = (unsigned short)(u >> 16);
    l = (unsigned short)(__float_as_uint(x - hf) >> 16);
}

// Pre-split a weight matrix into bf16 hi/lo planes (W is tiny & reused by
// every tile: doing this once keeps the main loop's VALU under the HBM roof).
__global__ void prep_w_split(const float* __restrict__ W,
                             unsigned short* __restrict__ hi,
                             unsigned short* __restrict__ lo, int n) {
    int i = blockIdx.x * 256 + threadIdx.x;
    if (i < n) {
        unsigned short h, l;
        split_bf16(W[i], h, l);
        hi[i] = h; lo[i] = l;
    }
}

// One wave computes a 16x128 output tile: 8 N-tiles of v_wmma_f32_16x16x32_bf16,
// 3 WMMAs per (k-chunk, n-tile) for the hi/lo mixed-precision product.
template<int DIN, bool PREP>
__global__ __launch_bounds__(256)
void hetero_linear(const float* __restrict__ X,            // [rows, DIN]
                   const float* __restrict__ W,            // [128, DIN]
                   const unsigned short* __restrict__ Whi, // [128*DIN] (PREP)
                   const unsigned short* __restrict__ Wlo, // [128*DIN] (PREP)
                   const float* __restrict__ bias,         // [128]
                   const int*   __restrict__ idx,          // [rows]
                   float*       __restrict__ out,          // [N, 128]
                   int ntiles) {
    const int lane = threadIdx.x & 31;
    const int wave = threadIdx.x >> 5;
    const int tile = blockIdx.x * 8 + wave;          // wave-uniform
    if (tile >= ntiles) return;                      // whole-wave exit: EXEC stays all-1s

    const int nl = lane & 15;        // A: row-in-tile for loads; B/C/D: column N
    const int kh = lane >> 4;        // K-half selector per ISA layouts
    const int m_base = tile * 16;

    const float* __restrict__ xrow = X + (size_t)(m_base + nl) * DIN;

    // Accumulators init with bias (bias depends only on column N = nl).
    v8f acc[8];
#pragma unroll
    for (int nt = 0; nt < 8; ++nt) {
        float bb = bias[nt * 16 + nl];
        v8f c;
#pragma unroll
        for (int i = 0; i < 8; ++i) c[i] = bb;
        acc[nt] = c;
    }

    for (int k0 = 0; k0 < DIN; k0 += 32) {
        // ---- A fragment (16x32, contiguous K order, same as B): lane holds row nl,
        // K = k0 + 16*kh + {0..15}.
        const float* ap = xrow + k0 + kh * 16;
        float4 a0 = *reinterpret_cast<const float4*>(ap);
        float4 a1 = *reinterpret_cast<const float4*>(ap + 4);
        float4 a2 = *reinterpret_cast<const float4*>(ap + 8);
        float4 a3 = *reinterpret_cast<const float4*>(ap + 12);
        float af[16] = {a0.x, a0.y, a0.z, a0.w, a1.x, a1.y, a1.z, a1.w,
                        a2.x, a2.y, a2.z, a2.w, a3.x, a3.y, a3.z, a3.w};
        v16us ah, al;
#pragma unroll
        for (int e = 0; e < 16; ++e) {
            unsigned short h, l;
            split_bf16(af[e], h, l);
            ah[e] = h; al[e] = l;
        }
        v16bf Ah = __builtin_bit_cast(v16bf, ah);
        v16bf Al = __builtin_bit_cast(v16bf, al);

#pragma unroll
        for (int nt = 0; nt < 8; ++nt) {
            // ---- B fragment (32x16): lane holds column n, K = k0 + 16*kh + {0..15}.
            v16bf Bh, Bl;
            const size_t woff = (size_t)(nt * 16 + nl) * DIN + k0 + kh * 16;
            if (PREP) {
                Bh = __builtin_bit_cast(v16bf, *reinterpret_cast<const v16us*>(Whi + woff));
                Bl = __builtin_bit_cast(v16bf, *reinterpret_cast<const v16us*>(Wlo + woff));
            } else {
                const float* wp = W + woff;
                float4 w0 = *reinterpret_cast<const float4*>(wp);
                float4 w1 = *reinterpret_cast<const float4*>(wp + 4);
                float4 w2 = *reinterpret_cast<const float4*>(wp + 8);
                float4 w3 = *reinterpret_cast<const float4*>(wp + 12);
                float wf[16] = {w0.x, w0.y, w0.z, w0.w, w1.x, w1.y, w1.z, w1.w,
                                w2.x, w2.y, w2.z, w2.w, w3.x, w3.y, w3.z, w3.w};
                v16us bh, bl;
#pragma unroll
                for (int e = 0; e < 16; ++e) {
                    unsigned short h, l;
                    split_bf16(wf[e], h, l);
                    bh[e] = h; bl[e] = l;
                }
                Bh = __builtin_bit_cast(v16bf, bh);
                Bl = __builtin_bit_cast(v16bf, bl);
            }
            // x*w ~= xh*wh + xh*wl + xl*wh  (drop xl*wl, ~2^-16 relative)
            acc[nt] = __builtin_amdgcn_wmma_f32_16x16x32_bf16(
                false, Ah, false, Bh, (short)0, acc[nt], false, false);
            acc[nt] = __builtin_amdgcn_wmma_f32_16x16x32_bf16(
                false, Ah, false, Bl, (short)0, acc[nt], false, false);
            acc[nt] = __builtin_amdgcn_wmma_f32_16x16x32_bf16(
                false, Al, false, Bh, (short)0, acc[nt], false, false);
        }
    }

    // ---- Scatter: C/D layout -> lane covers N = nl, VGPR r covers M = r + 8*kh.
    int rows[8];
#pragma unroll
    for (int r = 0; r < 8; ++r) rows[r] = idx[m_base + kh * 8 + r];
#pragma unroll
    for (int nt = 0; nt < 8; ++nt) {
#pragma unroll
        for (int r = 0; r < 8; ++r) {
            out[(size_t)rows[r] * 128 + nt * 16 + nl] = acc[nt][r];
        }
    }
}

extern "C" void kernel_launch(void* const* d_in, const int* in_sizes, int n_in,
                              void* d_out, int out_size, void* d_ws, size_t ws_size,
                              hipStream_t stream) {
    const float* x0   = (const float*)d_in[0];
    const float* x1   = (const float*)d_in[1];
    const float* x2   = (const float*)d_in[2];
    const float* W0   = (const float*)d_in[3];
    const float* b0   = (const float*)d_in[4];
    const float* W1   = (const float*)d_in[5];
    const float* b1   = (const float*)d_in[6];
    const float* W2   = (const float*)d_in[7];
    const float* b2   = (const float*)d_in[8];
    const int*   idx0 = (const int*)d_in[9];
    const int*   idx1 = (const int*)d_in[10];
    const int*   idx2 = (const int*)d_in[11];
    float* out = (float*)d_out;

    const int n0 = in_sizes[0] / 64;
    const int n1 = in_sizes[1] / 128;
    const int n2 = in_sizes[2] / 256;
    const int t0 = n0 / 16, t1 = n1 / 16, t2 = n2 / 16;   // M's are multiples of 16

    const size_t e0 = 128 * 64, e1 = 128 * 128, e2 = 128 * 256;
    const size_t need = 2 * (e0 + e1 + e2) * sizeof(unsigned short);  // ~229 KB
    const bool prep = (d_ws != nullptr) && (ws_size >= need);

    unsigned short* ws = (unsigned short*)d_ws;
    unsigned short* h0 = ws;
    unsigned short* l0 = h0 + e0;
    unsigned short* h1 = l0 + e0;
    unsigned short* l1 = h1 + e1;
    unsigned short* h2 = l1 + e1;
    unsigned short* l2 = h2 + e2;

    if (prep) {
        prep_w_split<<<(int)((e0 + 255) / 256), 256, 0, stream>>>(W0, h0, l0, (int)e0);
        prep_w_split<<<(int)((e1 + 255) / 256), 256, 0, stream>>>(W1, h1, l1, (int)e1);
        prep_w_split<<<(int)((e2 + 255) / 256), 256, 0, stream>>>(W2, h2, l2, (int)e2);
        hetero_linear<64,  true><<<(t0 + 7) / 8, 256, 0, stream>>>(x0, W0, h0, l0, b0, idx0, out, t0);
        hetero_linear<128, true><<<(t1 + 7) / 8, 256, 0, stream>>>(x1, W1, h1, l1, b1, idx1, out, t1);
        hetero_linear<256, true><<<(t2 + 7) / 8, 256, 0, stream>>>(x2, W2, h2, l2, b2, idx2, out, t2);
    } else {
        hetero_linear<64,  false><<<(t0 + 7) / 8, 256, 0, stream>>>(x0, W0, nullptr, nullptr, b0, idx0, out, t0);
        hetero_linear<128, false><<<(t1 + 7) / 8, 256, 0, stream>>>(x1, W1, nullptr, nullptr, b1, idx1, out, t1);
        hetero_linear<256, false><<<(t2 + 7) / 8, 256, 0, stream>>>(x2, W2, nullptr, nullptr, b2, idx2, out, t2);
    }
}
